// DND_SF_50010599195075
// MI455X (gfx1250) — matrix operand order repaired and checked by reading, not compiled, not run on hardware
//
#include <hip/hip_runtime.h>
#include <hip/hip_bf16.h>

typedef __attribute__((ext_vector_type(16))) _Float16 v16h;
typedef __attribute__((ext_vector_type(8)))  float    v8f;

// Problem constants (from reference)
#define NQ        1024
#define NKEYS     100000
#define DIM       512
#define KNN       32
// Tiling
#define TILE      128                 // keys per iteration (8 waves x 16)
#define NPAD      100096              // 782 * 128
#define ITERS     (NPAD / TILE)       // 782
#define CAND_MAX  2048                // worst-case pushes per iteration (8*32*8)
#define BIGF      3.0e38f

// ---------------------------------------------------------------------------
// Phase 1: dict_keys f32 -> f16 (Kh) + row squared norms (ksq). Pad rows get
// zero data and ksq = +BIG so they can never enter a top-k list.
// ---------------------------------------------------------------------------
__global__ __launch_bounds__(256)
void DND_prep_keys(const float* __restrict__ keys,
                   _Float16* __restrict__ kh,
                   float* __restrict__ ksq) {
    const int lane = threadIdx.x & 31;
    const int wave = threadIdx.x >> 5;
    const long long row = (long long)blockIdx.x * 8 + wave;
    if (row >= NPAD) return;

    float ss = 0.0f;
    _Float16* dst = kh + row * DIM;
    if (row < NKEYS) {
        const float* src = keys + row * DIM;
#pragma unroll
        for (int t = 0; t < 16; ++t) {
            float v = src[t * 32 + lane];
            ss = fmaf(v, v, ss);
            dst[t * 32 + lane] = (_Float16)v;
        }
    } else {
#pragma unroll
        for (int t = 0; t < 16; ++t) dst[t * 32 + lane] = (_Float16)0.0f;
    }
#pragma unroll
    for (int off = 16; off > 0; off >>= 1) ss += __shfl_xor(ss, off, 32);
    if (lane == 0) ksq[row] = (row < NKEYS) ? ss : BIGF;
}

// ---------------------------------------------------------------------------
// Phase 2+3 fused: WMMA score matmul + streaming top-32 + exact-distance
// weighted gather. One workgroup = 16 queries, 8 waves, 256 threads.
// B loads are software-pipelined 4 deep (incl. across iterations) and the
// K reduction uses two independent WMMA accumulator chains.
// ---------------------------------------------------------------------------
__global__ __launch_bounds__(256)
void DND_knn_main(const float* __restrict__ queries,
                  const float* __restrict__ dict_keys,
                  const float* __restrict__ dict_values,
                  const _Float16* __restrict__ kh,
                  const float* __restrict__ ksq,
                  float* __restrict__ out) {
    __shared__ float    qstage[16 * DIM];        // 32 KB, f32 query tile (also phase 3)
    __shared__ float    candD[2][CAND_MAX];      // 16 KB ping-pong candidate dists
    __shared__ unsigned candU[2][CAND_MAX];      // 16 KB packed (query<<20 | keyId)
    __shared__ unsigned candCnt[2];
    __shared__ float    thr[16];                 // per-query current 32nd-best
    __shared__ float    listD[16][KNN];
    __shared__ int      listI[16][KNN];

    const int tid  = threadIdx.x;
    const int lane = tid & 31;
    const int wave = tid >> 5;
    const int h    = lane >> 4;       // half-wave select (ISA A/B/D layouts)
    const int n    = lane & 15;       // N column within tile (B/D), M row (A)
    const int qBase = blockIdx.x * 16;

    // ---- stage query tile (f32) + init shared state ----
    for (int i = tid; i < 16 * DIM; i += 256) qstage[i] = queries[qBase * DIM + i];
    if (tid < 16) thr[tid] = BIGF;
    if (tid < 2)  candCnt[tid] = 0;
    for (int i = tid; i < 16 * KNN; i += 256) {
        (&listD[0][0])[i] = BIGF;
        (&listI[0][0])[i] = 0;
    }
    __syncthreads();

    // ---- build A fragments (16 queries x 512 dims) in registers: 16 k-steps ----
    // 16-bit A 16x32 layout (contiguous, matching B): lane (n) = M row; half j -> K = h*16+j.
    // 128 VGPRs, reused 782 times.
    v16h aFrag[16];
#pragma unroll
    for (int kk = 0; kk < 16; ++kk) {
#pragma unroll
        for (int j = 0; j < 16; ++j) {
            const int off = h * 16 + j;
            aFrag[kk][j] = (_Float16)qstage[n * DIM + kk * 32 + off];
        }
    }

    // merge-thread private state (only meaningful for tid < 16)
    float worstVal = BIGF;
    int   worstSlot = 0;

    const int bcol = h * 16;          // this lane's K-offset within a 32-dim slab

    // ---- prime the 4-deep B pipeline for iteration 0 ----
    const _Float16* bptr = kh + (long long)(wave * 16 + n) * DIM;
    v16h B0 = *(const v16h*)(bptr + 0 * 32 + bcol);
    v16h B1 = *(const v16h*)(bptr + 1 * 32 + bcol);
    v16h B2 = *(const v16h*)(bptr + 2 * 32 + bcol);
    v16h B3 = *(const v16h*)(bptr + 3 * 32 + bcol);

    // ---- main streaming loop over all keys ----
    for (int it = 0; it < ITERS; ++it) {
        const int p = it & 1;
        const int keyId = it * TILE + wave * 16 + n;
        const float ksqv = ksq[keyId];

        // vectorized threshold read for this lane's 8 query rows
        float thrl[8];
        {
            const float4* tp = (const float4*)&thr[h * 8];
            float4 ta = tp[0], tb = tp[1];
            thrl[0] = ta.x; thrl[1] = ta.y; thrl[2] = ta.z; thrl[3] = ta.w;
            thrl[4] = tb.x; thrl[5] = tb.y; thrl[6] = tb.z; thrl[7] = tb.w;
        }

        // next iteration's B base (last iter: any valid row; values unused)
        const _Float16* nbptr =
            kh + (long long)(((it + 1 < ITERS) ? (it + 1) * TILE : 0) + wave * 16 + n) * DIM;

        if (it + 1 < ITERS)   // keep L2 warm beyond the pipeline window
            __builtin_prefetch(nbptr + bcol, 0, 1);

        v8f c0 = {}, c1 = {};
#pragma unroll
        for (int kk = 0; kk < 16; kk += 2) {
            // issue loads 4 slabs ahead; last 2 steps preload next iteration
            const _Float16* p0 = (kk < 12) ? (bptr + (kk + 4) * 32 + bcol)
                                           : (nbptr + (kk - 12) * 32 + bcol);
            const _Float16* p1 = (kk < 12) ? (bptr + (kk + 5) * 32 + bcol)
                                           : (nbptr + (kk - 11) * 32 + bcol);
            v16h t0 = *(const v16h*)p0;
            v16h t1 = *(const v16h*)p1;
            if ((kk & 3) == 0) {
                c0 = __builtin_amdgcn_wmma_f32_16x16x32_f16(
                         false, aFrag[kk],     false, B0, (short)0, c0, false, false);
                c1 = __builtin_amdgcn_wmma_f32_16x16x32_f16(
                         false, aFrag[kk + 1], false, B1, (short)0, c1, false, false);
                B0 = t0; B1 = t1;
            } else {
                c0 = __builtin_amdgcn_wmma_f32_16x16x32_f16(
                         false, aFrag[kk],     false, B2, (short)0, c0, false, false);
                c1 = __builtin_amdgcn_wmma_f32_16x16x32_f16(
                         false, aFrag[kk + 1], false, B3, (short)0, c1, false, false);
                B2 = t0; B3 = t1;
            }
        }
        bptr = nbptr;

        // D layout: VGPR r -> M = r + 8h, lane -> N.  Rank by ksq - 2*(q.k).
#pragma unroll
        for (int r = 0; r < 8; ++r) {
            const float d = fmaf(-2.0f, c0[r] + c1[r], ksqv);
            if (d < thrl[r]) {
                const int mq = r + 8 * h;
                unsigned pos = atomicAdd(&candCnt[p], 1u);
                if (pos < CAND_MAX) {
                    candD[p][pos] = d;
                    candU[p][pos] = ((unsigned)mq << 20) | (unsigned)keyId;
                }
            }
        }
        __syncthreads();

        // ---- serial merge: thread q owns query q's list (rare after warm-up)
        if (tid < 16) {
            unsigned cN = candCnt[p];
            if (cN > CAND_MAX) cN = CAND_MAX;
            for (unsigned i = 0; i < cN; ++i) {
                const unsigned u = candU[p][i];
                if ((u >> 20) == (unsigned)tid) {
                    const float d = candD[p][i];
                    if (d < worstVal) {
                        listD[tid][worstSlot] = d;
                        listI[tid][worstSlot] = (int)(u & 0xFFFFFu);
                        float wv = -BIGF; int ws = 0;
#pragma unroll
                        for (int s = 0; s < KNN; ++s) {
                            const float v = listD[tid][s];
                            if (v > wv) { wv = v; ws = s; }
                        }
                        worstVal = wv; worstSlot = ws;
                    }
                }
            }
            thr[tid] = worstVal;
        } else if (tid == 16) {
            candCnt[1 - p] = 0;   // safe: last read of that counter was 2 barriers ago
        }
        __syncthreads();
    }

    // ---- phase 3: exact f32 distances on the 32 neighbours + weighted values
    // Each wave handles 2 queries; dims mapped d = t*32 + lane (coalesced).
#pragma unroll 1
    for (int sub = 0; sub < 2; ++sub) {
        const int q = wave * 2 + sub;
        float qv[16], acc[16];
        float wsum = 0.0f;
#pragma unroll
        for (int t = 0; t < 16; ++t) {
            qv[t] = qstage[q * DIM + t * 32 + lane];
            acc[t] = 0.0f;
        }
        for (int nb = 0; nb < KNN; ++nb) {
            const int idx = listI[q][nb];
            const float* krow = dict_keys + (long long)idx * DIM;
            float dl = 0.0f;
#pragma unroll
            for (int t = 0; t < 16; ++t) {
                const float df = qv[t] - krow[t * 32 + lane];
                dl = fmaf(df, df, dl);
            }
#pragma unroll
            for (int off = 16; off > 0; off >>= 1) dl += __shfl_xor(dl, off, 32);
            const float w = 1.0f / (dl + 1.0e-3f);
            wsum += w;
            const float* vrow = dict_values + (long long)idx * DIM;
#pragma unroll
            for (int t = 0; t < 16; ++t)
                acc[t] = fmaf(w, vrow[t * 32 + lane], acc[t]);
        }
        const float inv = 1.0f / wsum;
#pragma unroll
        for (int t = 0; t < 16; ++t)
            out[(long long)(qBase + q) * DIM + t * 32 + lane] = acc[t] * inv;
    }
}

// ---------------------------------------------------------------------------
extern "C" void kernel_launch(void* const* d_in, const int* in_sizes, int n_in,
                              void* d_out, int out_size, void* d_ws, size_t ws_size,
                              hipStream_t stream) {
    const float* queries     = (const float*)d_in[0];   // [1024, 512]
    const float* dict_keys   = (const float*)d_in[1];   // [100000, 512]
    const float* dict_values = (const float*)d_in[2];   // [100000, 512]
    float*       out         = (float*)d_out;           // [1024, 512]

    // workspace layout: Kh (padded f16 keys) then ksq (padded f32 norms)
    const size_t KH_BYTES = (size_t)NPAD * DIM * sizeof(_Float16);  // ~102.5 MB
    _Float16* kh  = (_Float16*)d_ws;
    float*    ksq = (float*)((char*)d_ws + KH_BYTES);

    // Phase 1: convert + norms (8 rows per 256-thread block)
    DND_prep_keys<<<NPAD / 8, 256, 0, stream>>>(dict_keys, kh, ksq);

    // Phase 2+3: fused WMMA scores + top-32 + weighted gather
    DND_knn_main<<<NQ / 16, 256, 0, stream>>>(queries, dict_keys, dict_values,
                                              kh, ksq, out);
}
